// DecoderBlock_38963943309859
// MI455X (gfx1250) — compile-verified, matrix operand order checked by reading, not run on hardware
//
#include <hip/hip_runtime.h>
#include <cstdint>
#include <cstddef>

// ---------------- problem constants (match reference) ----------------
#define EMB   1024
#define HEADS 16
#define HDIM  64
#define TSEQ  2048
#define BATCH 2
#define BT    (BATCH * TSEQ)   // 4096 rows
#define FF    (4 * EMB)        // 4096

typedef __attribute__((ext_vector_type(16))) __bf16 v16bf;
typedef __attribute__((ext_vector_type(8)))  float  v8f;
typedef unsigned short u16;

union Frag16 { v16bf v; uint4 q[2]; };
union Pack8  { uint4 q; u16 s[8]; };

__device__ __forceinline__ u16 f2bf(float f) {
  unsigned u = __float_as_uint(f);
  u += 0x7FFFu + ((u >> 16) & 1u);   // round-to-nearest-even
  return (u16)(u >> 16);
}

__device__ __forceinline__ void storeval(float* p, float v) { *p = v; }
__device__ __forceinline__ void storeval(u16* p, float v)   { *p = f2bf(v); }

// ---- CDNA5 async global->LDS copy (ASYNCcnt-tracked), inline asm ----
__device__ __forceinline__ void async_ld128(const u16* g, u16* lds) {
  unsigned loff = (unsigned)(size_t)lds;   // LDS aperture: low 32 bits = LDS addr
  asm volatile("global_load_async_to_lds_b128 %0, %1, off"
               :: "v"(loff), "v"(g) : "memory");
}
__device__ __forceinline__ void wait_async0() {
  asm volatile("s_wait_asynccnt 0" ::: "memory");
}

// ---------------- weight packing ----------------
// wq/wk/wv: [H, EMB, HD] f32 -> W [EMB, 3*EMB] bf16 (K-major, n = which*1024 + h*64 + d)
__global__ __launch_bounds__(256) void pack_qkv_kernel(
    const float* __restrict__ wq, const float* __restrict__ wk,
    const float* __restrict__ wv, u16* __restrict__ W) {
  size_t idx = (size_t)blockIdx.x * 256 + threadIdx.x;       // < 3*1024*1024
  int which = (int)(idx >> 20);
  int r     = (int)(idx & 0xFFFFF);
  int c     = r >> 10;           // K index (emb)
  int col   = r & 1023;          // h*64+d
  int h     = col >> 6;
  int d     = col & 63;
  const float* src = (which == 0) ? wq : (which == 1) ? wk : wv;
  float v = src[(size_t)h * EMB * HDIM + (size_t)c * HDIM + d];
  W[(size_t)c * (3 * EMB) + (size_t)which * EMB + col] = f2bf(v);
}

__global__ __launch_bounds__(256) void cvt_bf16_kernel(
    const float* __restrict__ src, u16* __restrict__ dst, size_t n) {
  size_t i = (size_t)blockIdx.x * 256 + threadIdx.x;
  if (i < n) dst[i] = f2bf(src[i]);
}

// ---------------- layernorm: f32 row -> bf16 row ----------------
__global__ __launch_bounds__(256) void ln_bf16_kernel(
    const float* __restrict__ x, const float* __restrict__ g,
    const float* __restrict__ b, u16* __restrict__ out) {
  const int row = blockIdx.x;
  const int tid = threadIdx.x;
  const int wid = tid >> 5;
  const float* xr = x + (size_t)row * EMB;
  float4 xv = *(const float4*)&xr[tid * 4];
  float s  = xv.x + xv.y + xv.z + xv.w;
  float sq = xv.x * xv.x + xv.y * xv.y + xv.z * xv.z + xv.w * xv.w;
  for (int d = 16; d > 0; d >>= 1) {
    s  += __shfl_xor(s,  d, 32);
    sq += __shfl_xor(sq, d, 32);
  }
  __shared__ float red[16];
  if ((tid & 31) == 0) { red[wid] = s; red[8 + wid] = sq; }
  __syncthreads();
  float ts = 0.f, tq = 0.f;
#pragma unroll
  for (int i = 0; i < 8; ++i) { ts += red[i]; tq += red[8 + i]; }
  float mean = ts * (1.f / EMB);
  float var  = tq * (1.f / EMB) - mean * mean;
  float rstd = rsqrtf(var + 1e-5f);
  float4 gv = *(const float4*)&g[tid * 4];
  float4 bv = *(const float4*)&b[tid * 4];
  u16 o0 = f2bf((xv.x - mean) * rstd * gv.x + bv.x);
  u16 o1 = f2bf((xv.y - mean) * rstd * gv.y + bv.y);
  u16 o2 = f2bf((xv.z - mean) * rstd * gv.z + bv.z);
  u16 o3 = f2bf((xv.w - mean) * rstd * gv.w + bv.w);
  uint2 ov; ov.x = (unsigned)o0 | ((unsigned)o1 << 16);
  ov.y = (unsigned)o2 | ((unsigned)o3 << 16);
  *(uint2*)&out[(size_t)row * EMB + tid * 4] = ov;
}

// ---------------- tiled bf16 GEMM with WMMA + async LDS staging ----------------
// C[M,N] = A[M,K] * B[K,N] (+bias, +ReLU, +residual). 128x128 block, BK=32.
// 8 waves: wave (wm,wn) in 4x2 owns a 32x64 sub-tile = 2x4 wmma accumulators.
// A tile staged with global_load_async_to_lds_b128; B tile transposed via regs.
template <typename OutT, bool RELU>
__global__ __launch_bounds__(256) void gemm_bf16_kernel(
    const u16* __restrict__ A, const u16* __restrict__ Bw,
    const float* __restrict__ bias, const float* res,
    OutT* out, int M, int N, int K) {
  __shared__ u16 As[2][128 * 32];   // row-major [m][k]
  __shared__ u16 Bs[2][128 * 32];   // n-major  [n][k]
  const int tid  = threadIdx.x;
  const int wid  = tid >> 5;
  const int ln   = tid & 31;
  const int l16  = ln & 15;
  const int half = (ln >> 4) << 3;           // 0 or 8 (K-halves per 16-bit WMMA layout)
  const int wm   = wid >> 1;
  const int wn   = wid & 1;
  const int m0   = blockIdx.y * 128;
  const int n0   = blockIdx.x * 128;
  const int nk   = K >> 5;

  v8f acc[2][4];
  const v8f vzero = {0.f, 0.f, 0.f, 0.f, 0.f, 0.f, 0.f, 0.f};
#pragma unroll
  for (int mt = 0; mt < 2; ++mt)
#pragma unroll
    for (int nt = 0; nt < 4; ++nt) acc[mt][nt] = vzero;

  // per-thread staging chunks (2 x 16B each for A and B)
  const int ea0 = tid * 8, ea1 = (tid + 256) * 8;
  const int arow0 = ea0 >> 5, acol0 = ea0 & 31;
  const int arow1 = ea1 >> 5, acol1 = ea1 & 31;
  const int bk0 = ea0 >> 7, bn0 = ea0 & 127;
  const int bk1 = ea1 >> 7, bn1 = ea1 & 127;

  // prologue: k-block 0
  async_ld128(&A[(size_t)(m0 + arow0) * K + acol0], &As[0][ea0]);
  async_ld128(&A[(size_t)(m0 + arow1) * K + acol1], &As[0][ea1]);
  uint4 br0 = *(const uint4*)&Bw[(size_t)bk0 * N + n0 + bn0];
  uint4 br1 = *(const uint4*)&Bw[(size_t)bk1 * N + n0 + bn1];
  { Pack8 p; p.q = br0;
#pragma unroll
    for (int t = 0; t < 8; ++t) Bs[0][(bn0 + t) * 32 + bk0] = p.s[t]; }
  { Pack8 p; p.q = br1;
#pragma unroll
    for (int t = 0; t < 8; ++t) Bs[0][(bn1 + t) * 32 + bk1] = p.s[t]; }

  for (int kb = 0; kb < nk; ++kb) {
    wait_async0();        // this wave's async LDS writes for As[cur] done
    __syncthreads();      // everyone's tile (async A + ds B) visible
    const int cur = kb & 1, nxt = cur ^ 1;
    if (kb + 2 < nk) {    // L2 prefetch hints two k-blocks ahead (B path)
      const int kp = (kb + 2) * 32;
      __builtin_prefetch(&Bw[(size_t)(kp + bk0) * N + n0 + bn0], 0, 1);
    }
    if (kb + 1 < nk) {
      const int ko = (kb + 1) * 32;
      async_ld128(&A[(size_t)(m0 + arow0) * K + ko + acol0], &As[nxt][ea0]);
      async_ld128(&A[(size_t)(m0 + arow1) * K + ko + acol1], &As[nxt][ea1]);
      br0 = *(const uint4*)&Bw[(size_t)(ko + bk0) * N + n0 + bn0];
      br1 = *(const uint4*)&Bw[(size_t)(ko + bk1) * N + n0 + bn1];
    }
    Frag16 af[2], bfr[4];
#pragma unroll
    for (int mt = 0; mt < 2; ++mt) {
      const int base = (wm * 32 + mt * 16 + l16) * 32 + half;
      af[mt].q[0] = *(const uint4*)&As[cur][base];
      af[mt].q[1] = *(const uint4*)&As[cur][base + 16];
    }
#pragma unroll
    for (int nt = 0; nt < 4; ++nt) {
      const int base = (wn * 64 + nt * 16 + l16) * 32 + half;
      bfr[nt].q[0] = *(const uint4*)&Bs[cur][base];
      bfr[nt].q[1] = *(const uint4*)&Bs[cur][base + 16];
    }
#pragma unroll
    for (int mt = 0; mt < 2; ++mt)
#pragma unroll
      for (int nt = 0; nt < 4; ++nt)
        acc[mt][nt] = __builtin_amdgcn_wmma_f32_16x16x32_bf16(
            false, af[mt].v, false, bfr[nt].v, (short)0, acc[mt][nt], false, false);
    if (kb + 1 < nk) {
      { Pack8 p; p.q = br0;
#pragma unroll
        for (int t = 0; t < 8; ++t) Bs[nxt][(bn0 + t) * 32 + bk0] = p.s[t]; }
      { Pack8 p; p.q = br1;
#pragma unroll
        for (int t = 0; t < 8; ++t) Bs[nxt][(bn1 + t) * 32 + bk1] = p.s[t]; }
    }
  }

  // epilogue: C/D layout row = i + half, col = l16
#pragma unroll
  for (int mt = 0; mt < 2; ++mt) {
#pragma unroll
    for (int nt = 0; nt < 4; ++nt) {
      const int gn = n0 + wn * 64 + nt * 16 + l16;
#pragma unroll
      for (int i = 0; i < 8; ++i) {
        const int gm = m0 + wm * 32 + mt * 16 + half + i;
        float v = acc[mt][nt][i];
        if (bias) v += bias[gn];
        if (RELU) v = fmaxf(v, 0.f);
        if (res)  v += res[(size_t)gm * N + gn];
        storeval(&out[(size_t)gm * N + gn], v);
      }
    }
  }
}

// ---------------- flash attention (causal, scale = EMB^-0.5) ----------------
// qkv: [BT, 3*EMB] bf16 (cols: q=h*64+d, k=+1024, v=+2048) -> ctx [BT, EMB] bf16
__global__ __launch_bounds__(256) void attn_kernel(
    const u16* __restrict__ qkv, u16* __restrict__ ctx) {
  __shared__ u16 Ks[64 * 64];      // [key][hd] row-major (== B layout for Q*K^T)
  __shared__ u16 Vt[64 * 64];      // [hd][key] transposed (== B layout for P*V)
  __shared__ u16 Pbuf[8 * 16 * 64];
  const int tid  = threadIdx.x;
  const int wid  = tid >> 5;
  const int ln   = tid & 31;
  const int l16  = ln & 15;
  const int half = (ln >> 4) << 3;
  const int bh = blockIdx.y;
  const int b  = bh >> 4;
  const int h  = bh & 15;
  const int m0 = blockIdx.x * 128;
  const int r0 = m0 + wid * 16;                 // this wave's 16 query rows
  const size_t rowS = 3 * EMB;
  const size_t base_bh = (size_t)b * TSEQ * rowS + (size_t)h * HDIM;

  // Q fragments stay in registers the whole kernel (A layout: row = l16)
  Frag16 qf[2];
  const size_t qoff = base_bh + (size_t)(r0 + l16) * rowS;
#pragma unroll
  for (int c = 0; c < 2; ++c) {
    qf[c].q[0] = *(const uint4*)&qkv[qoff + c * 32 + half];
    qf[c].q[1] = *(const uint4*)&qkv[qoff + c * 32 + half + 16];
  }

  float m_i[8], l_i[8];
  v8f ov[4];
  const v8f vzero = {0.f, 0.f, 0.f, 0.f, 0.f, 0.f, 0.f, 0.f};
#pragma unroll
  for (int i = 0; i < 8; ++i) { m_i[i] = -__builtin_inff(); l_i[i] = 0.f; }
#pragma unroll
  for (int nt = 0; nt < 4; ++nt) ov[nt] = vzero;

  const float scale = 0.03125f;   // 1024^-0.5 (reference scales by full emb)
  const int ntiles = (m0 >> 6) + 2;

  for (int j = 0; j < ntiles; ++j) {
    const int k0 = j * 64;
    // stage K (async, row-major) and V (transposed via regs)
#pragma unroll
    for (int i = 0; i < 2; ++i) {
      const int e  = (tid + i * 256) * 8;
      const int kr = e >> 6, kc = e & 63;
      const size_t g = base_bh + (size_t)(k0 + kr) * rowS + kc;
      async_ld128(&qkv[g + EMB], &Ks[e]);                   // K
      Pack8 p; p.q = *(const uint4*)&qkv[g + 2 * EMB];      // V
#pragma unroll
      for (int t = 0; t < 8; ++t) Vt[(kc + t) * 64 + kr] = p.s[t];
    }
    wait_async0();
    __syncthreads();

    // S = Q @ K^T   (16 x 64 per wave)
    v8f s[4];
#pragma unroll
    for (int nt = 0; nt < 4; ++nt) s[nt] = vzero;
#pragma unroll
    for (int c = 0; c < 2; ++c) {
      Frag16 kf[4];
#pragma unroll
      for (int nt = 0; nt < 4; ++nt) {
        const int base = (nt * 16 + l16) * 64 + c * 32 + half;
        kf[nt].q[0] = *(const uint4*)&Ks[base];
        kf[nt].q[1] = *(const uint4*)&Ks[base + 16];
      }
#pragma unroll
      for (int nt = 0; nt < 4; ++nt)
        s[nt] = __builtin_amdgcn_wmma_f32_16x16x32_bf16(
            false, qf[c].v, false, kf[nt].v, (short)0, s[nt], false, false);
    }

    // scale + causal mask
#pragma unroll
    for (int nt = 0; nt < 4; ++nt) {
      const int kcol = k0 + nt * 16 + l16;
#pragma unroll
      for (int i = 0; i < 8; ++i) {
        const int qr = r0 + half + i;
        float v = s[nt][i] * scale;
        if (kcol > qr) v = -__builtin_inff();
        s[nt][i] = v;
      }
    }

    // online softmax per row (rows live across the 16 lanes of a half-wave)
    float alpha[8];
#pragma unroll
    for (int i = 0; i < 8; ++i) {
      float mx = s[0][i];
#pragma unroll
      for (int nt = 1; nt < 4; ++nt) mx = fmaxf(mx, s[nt][i]);
      for (int d = 1; d < 16; d <<= 1) mx = fmaxf(mx, __shfl_xor(mx, d, 32));
      float mnew = fmaxf(m_i[i], mx);
      alpha[i] = expf(m_i[i] - mnew);
      m_i[i] = mnew;
      float sum = 0.f;
#pragma unroll
      for (int nt = 0; nt < 4; ++nt) {
        float p = expf(s[nt][i] - mnew);
        s[nt][i] = p;
        sum += p;
      }
      for (int d = 1; d < 16; d <<= 1) sum += __shfl_xor(sum, d, 32);
      l_i[i] = l_i[i] * alpha[i] + sum;
    }
#pragma unroll
    for (int nt = 0; nt < 4; ++nt)
#pragma unroll
      for (int i = 0; i < 8; ++i) ov[nt][i] *= alpha[i];

    // re-fragment P through per-wave LDS (C/D layout -> A layout)
    const int pb = wid * 1024;
#pragma unroll
    for (int nt = 0; nt < 4; ++nt)
#pragma unroll
      for (int i = 0; i < 8; ++i)
        Pbuf[pb + (half + i) * 64 + nt * 16 + l16] = f2bf(s[nt][i]);

    // O += P @ V
#pragma unroll
    for (int c2 = 0; c2 < 2; ++c2) {
      Frag16 pf;
      const int pbase = pb + l16 * 64 + c2 * 32 + half;
      pf.q[0] = *(const uint4*)&Pbuf[pbase];
      pf.q[1] = *(const uint4*)&Pbuf[pbase + 16];
#pragma unroll
      for (int nt = 0; nt < 4; ++nt) {
        Frag16 vf;
        const int vb = (nt * 16 + l16) * 64 + c2 * 32 + half;
        vf.q[0] = *(const uint4*)&Vt[vb];
        vf.q[1] = *(const uint4*)&Vt[vb + 16];
        ov[nt] = __builtin_amdgcn_wmma_f32_16x16x32_bf16(
            false, pf.v, false, vf.v, (short)0, ov[nt], false, false);
      }
    }
    __syncthreads();   // before Ks/Vt are overwritten
  }

  // normalize and write context (concat-head layout)
#pragma unroll
  for (int i = 0; i < 8; ++i) {
    const float inv = 1.f / l_i[i];
    const size_t orow = ((size_t)b * TSEQ + r0 + half + i) * EMB + (size_t)h * HDIM;
#pragma unroll
    for (int nt = 0; nt < 4; ++nt)
      ctx[orow + nt * 16 + l16] = f2bf(ov[nt][i] * inv);
  }
}

// ---------------- launcher ----------------
extern "C" void kernel_launch(void* const* d_in, const int* in_sizes, int n_in,
                              void* d_out, int out_size, void* d_ws, size_t ws_size,
                              hipStream_t stream) {
  (void)in_sizes; (void)n_in; (void)out_size; (void)ws_size;
  const float* x   = (const float*)d_in[0];
  const float* wq  = (const float*)d_in[1];
  const float* wk  = (const float*)d_in[2];
  const float* wv  = (const float*)d_in[3];
  const float* wo  = (const float*)d_in[4];
  const float* bo  = (const float*)d_in[5];
  const float* g1  = (const float*)d_in[6];
  const float* b1  = (const float*)d_in[7];
  const float* g2  = (const float*)d_in[8];
  const float* b2  = (const float*)d_in[9];
  const float* w1  = (const float*)d_in[10];
  const float* bf1 = (const float*)d_in[11];
  const float* w2  = (const float*)d_in[12];
  const float* bf2 = (const float*)d_in[13];
  float* out = (float*)d_out;

  char* ws = (char*)d_ws;
  const size_t OFF_WQKV = 0;                                          // 6 MB
  const size_t OFF_WO   = OFF_WQKV + (size_t)3 * EMB * EMB * 2;       // +2 MB
  const size_t OFF_W1   = OFF_WO   + (size_t)EMB * EMB * 2;           // +8 MB
  const size_t OFF_W2   = OFF_W1   + (size_t)EMB * FF * 2;            // +8 MB
  const size_t OFF_HA   = OFF_W2   + (size_t)FF * EMB * 2;            // +8 MB
  const size_t OFF_QKV  = OFF_HA   + (size_t)BT * EMB * 2;            // +24 MB
  const size_t OFF_CTX  = OFF_QKV  + (size_t)BT * 3 * EMB * 2;        // +8 MB
  const size_t OFF_H2   = OFF_CTX  + (size_t)BT * EMB * 2;            // +8 MB
  const size_t OFF_F1   = OFF_H2   + (size_t)BT * EMB * 2;            // +32 MB
  u16* Wqkv = (u16*)(ws + OFF_WQKV);
  u16* Wo   = (u16*)(ws + OFF_WO);
  u16* W1   = (u16*)(ws + OFF_W1);
  u16* W2   = (u16*)(ws + OFF_W2);
  u16* hA   = (u16*)(ws + OFF_HA);
  u16* QKV  = (u16*)(ws + OFF_QKV);
  u16* CTX  = (u16*)(ws + OFF_CTX);
  u16* H2   = (u16*)(ws + OFF_H2);
  u16* F1   = (u16*)(ws + OFF_F1);

  // weight packing (bf16)
  pack_qkv_kernel<<<(3 * EMB * EMB) / 256, 256, 0, stream>>>(wq, wk, wv, Wqkv);
  cvt_bf16_kernel<<<(EMB * EMB) / 256, 256, 0, stream>>>(wo, Wo, (size_t)EMB * EMB);
  cvt_bf16_kernel<<<(EMB * FF) / 256, 256, 0, stream>>>(w1, W1, (size_t)EMB * FF);
  cvt_bf16_kernel<<<(FF * EMB) / 256, 256, 0, stream>>>(w2, W2, (size_t)FF * EMB);

  // LN1
  ln_bf16_kernel<<<BT, 256, 0, stream>>>(x, g1, b1, hA);
  // QKV = h @ Wqkv   (4096 x 3072)
  gemm_bf16_kernel<u16, false><<<dim3(3 * EMB / 128, BT / 128), 256, 0, stream>>>(
      hA, Wqkv, nullptr, nullptr, QKV, BT, 3 * EMB, EMB);
  // attention
  attn_kernel<<<dim3(TSEQ / 128, BATCH * HEADS), 256, 0, stream>>>(QKV, CTX);
  // x1 = x + ctx @ Wo + bo   -> d_out (f32)
  gemm_bf16_kernel<float, false><<<dim3(EMB / 128, BT / 128), 256, 0, stream>>>(
      CTX, Wo, bo, x, out, BT, EMB, EMB);
  // LN2 on x1
  ln_bf16_kernel<<<BT, 256, 0, stream>>>(out, g2, b2, H2);
  // f1 = relu(h2 @ W1 + bf1)
  gemm_bf16_kernel<u16, true><<<dim3(FF / 128, BT / 128), 256, 0, stream>>>(
      H2, W1, bf1, nullptr, F1, BT, FF, EMB);
  // out = x1 + f1 @ W2 + bf2
  gemm_bf16_kernel<float, false><<<dim3(EMB / 128, BT / 128), 256, 0, stream>>>(
      F1, W2, bf2, out, out, BT, EMB, FF);
}
